// EncoderLayer_71193377899272
// MI455X (gfx1250) — matrix operand order repaired and checked by reading, not compiled, not run on hardware
//
#include <hip/hip_runtime.h>
#include <hip/hip_bf16.h>

typedef unsigned short u16;
typedef __attribute__((ext_vector_type(16))) __bf16 v16bf;
typedef __attribute__((ext_vector_type(8)))  __bf16 v8bf;
typedef __attribute__((ext_vector_type(8)))  float  v8f;

// Problem constants (match reference)
static constexpr int kB = 2, kS = 2048, kD = 1024, kH = 16, kF = 4096, kE = 8;
static constexpr int kM = kB * kS;          // 4096 tokens
static constexpr int kDK = kD / kH;         // 64

// Epilogue modes
enum { EPI_PHI = 0, EPI_GF32 = 1, EPI_GBF16 = 2, EPI_RESID = 3, EPI_SACC = 4 };

// CDNA5 async global->LDS path (ASYNCcnt-tracked). Guarded so either toolchain compiles.
#if __has_builtin(__builtin_amdgcn_global_load_async_to_lds_b128) && \
    __has_builtin(__builtin_amdgcn_s_wait_asynccnt)
#define USE_ASYNC_LDS 1
#else
#define USE_ASYNC_LDS 0
#endif

#if USE_ASYNC_LDS
typedef int v4i_ __attribute__((vector_size(16)));
typedef __attribute__((address_space(1))) v4i_* as1_v4i;   // global (AS1) int4*
typedef __attribute__((address_space(3))) v4i_* as3_v4i;   // LDS (AS3) int4*
// Global flat address == AS1 address; LDS flat address low 32 bits == DS address
// (ISA 10.2 aperture mapping), and AS3 pointers are 32-bit on amdgcn.
__device__ inline void async_b128(const u16* g, u16* l) {
    __builtin_amdgcn_global_load_async_to_lds_b128(
        (as1_v4i)(uintptr_t)g,
        (as3_v4i)(unsigned)(uintptr_t)l, 0, 0);
}
#endif

__device__ inline u16 f2bf(float f) {
    unsigned u = __float_as_uint(f);
    unsigned r = u + 0x7FFFu + ((u >> 16) & 1u);   // round-to-nearest-even
    return (u16)(r >> 16);
}
__device__ inline float bf2f(u16 h) { return __uint_as_float(((unsigned)h) << 16); }

__device__ inline v16bf load16bf(const u16* p0, const u16* p1) {
    v8bf lo = *(const v8bf*)(const void*)p0;
    v8bf hi = *(const v8bf*)(const void*)p1;
    return __builtin_shufflevector(lo, hi, 0,1,2,3,4,5,6,7,8,9,10,11,12,13,14,15);
}

// ---------------------------------------------------------------------------
// Pack: fp32 W[K][N] -> bf16 W^T[N][K] (LDS tile transpose, coalesced both ways)
// ---------------------------------------------------------------------------
__global__ __launch_bounds__(256) void packT(const float* __restrict__ in,
                                             u16* __restrict__ out, int K, int N) {
    __shared__ float tile[32][33];
    int nb = blockIdx.x * 32, kb = blockIdx.y * 32;
    int tx = threadIdx.x & 31, ty = threadIdx.x >> 5;      // ty: 0..7
    #pragma unroll
    for (int r = 0; r < 32; r += 8)
        tile[r + ty][tx] = in[(size_t)(kb + r + ty) * N + nb + tx];
    __syncthreads();
    #pragma unroll
    for (int r = 0; r < 32; r += 8)
        out[(size_t)(nb + r + ty) * K + kb + tx] = f2bf(tile[tx][r + ty]);
}

// ---------------------------------------------------------------------------
// LayerNorm over D=1024, one row per block, bf16 output
// ---------------------------------------------------------------------------
__global__ __launch_bounds__(256) void ln_kernel(const float* __restrict__ x,
                                                 const float* __restrict__ g,
                                                 const float* __restrict__ b,
                                                 u16* __restrict__ out, float eps) {
    __shared__ float s1[256], s2[256];
    int row = blockIdx.x, t = threadIdx.x;
    const float* xr = x + (size_t)row * kD;
    float v[4], sum = 0.f, sq = 0.f;
    #pragma unroll
    for (int i = 0; i < 4; i++) { v[i] = xr[t + i * 256]; sum += v[i]; sq += v[i] * v[i]; }
    s1[t] = sum; s2[t] = sq; __syncthreads();
    for (int st = 128; st > 0; st >>= 1) {
        if (t < st) { s1[t] += s1[t + st]; s2[t] += s2[t + st]; }
        __syncthreads();
    }
    float m = s1[0] * (1.f / kD);
    float inv = rsqrtf(s2[0] * (1.f / kD) - m * m + eps);
    #pragma unroll
    for (int i = 0; i < 4; i++) {
        int c = t + i * 256;
        out[(size_t)row * kD + c] = f2bf((v[i] - m) * inv * g[c] + b[c]);
    }
}

// ---------------------------------------------------------------------------
// Core WMMA GEMM: C[M,N] = A[M,K](bf16) @ B1T[N,K](bf16)  (+ gated second GEMM)
// Workgroup: 8 waves -> 128(M) x 64(N) block; K-step 32.
// B panels double-buffered in LDS; staged with GLOBAL_LOAD_ASYNC_TO_LDS_B128
// (ASYNCcnt) when available; B fragments pre-loaded so WMMAs issue as a burst.
// ---------------------------------------------------------------------------
template <bool GATED>
__global__ __launch_bounds__(256) void gemm_wmma(
    const u16* __restrict__ A, int Kdim, int N,
    const u16* __restrict__ B1T, const float* __restrict__ bias1,
    const u16* __restrict__ B2T, const float* __restrict__ bias2,
    float* __restrict__ outF, u16* __restrict__ outB,
    const float* __restrict__ resid, const float* __restrict__ scale,
    int scaleStride, int epi)
{
    constexpr int NB = GATED ? 2 : 1;
    __shared__ u16 lds[2][NB][64 * 32];        // double-buffered B panels

    const int lane  = threadIdx.x & 31;
    const int wave  = threadIdx.x >> 5;
    const int nBase = blockIdx.x * 64;
    const int mBase = blockIdx.y * 128;
    const int mRow  = mBase + wave * 16 + (lane & 15);
    const int kHalf = (lane >> 4) * 16;    // A: lanes 0-15 K 0..15, lanes 16-31 K 16..31

    v8f acc1[4], acc2[4];
    const v8f zero = {0.f,0.f,0.f,0.f,0.f,0.f,0.f,0.f};
    #pragma unroll
    for (int i = 0; i < 4; i++) { acc1[i] = zero; if constexpr (GATED) acc2[i] = zero; }

    const int lcol = threadIdx.x >> 2;           // 0..63 cols of B panel
    const int lseg = (threadIdx.x & 3) * 8;      // 8 bf16 (16B) segments
    const int nIter = Kdim / 32;

    auto stage = [&](int iter, int buf) {
        const int k0 = iter * 32;
        const u16* g1 = B1T + (size_t)(nBase + lcol) * Kdim + k0 + lseg;
#if USE_ASYNC_LDS
        async_b128(g1, &lds[buf][0][lcol * 32 + lseg]);
        if constexpr (GATED) {
            const u16* g2 = B2T + (size_t)(nBase + lcol) * Kdim + k0 + lseg;
            async_b128(g2, &lds[buf][1][lcol * 32 + lseg]);
        }
#else
        *(uint4*)&lds[buf][0][lcol * 32 + lseg] = *(const uint4*)g1;
        __builtin_prefetch(g1 + 32, 0, 1);
        if constexpr (GATED) {
            const u16* g2 = B2T + (size_t)(nBase + lcol) * Kdim + k0 + lseg;
            *(uint4*)&lds[buf][1][lcol * 32 + lseg] = *(const uint4*)g2;
            __builtin_prefetch(g2 + 32, 0, 1);
        }
#endif
    };

    stage(0, 0);
    for (int it = 0; it < nIter; ++it) {
        const int buf = it & 1;
        if (it + 1 < nIter) {
            stage(it + 1, buf ^ 1);              // overlap next panel with compute
#if USE_ASYNC_LDS
            __builtin_amdgcn_s_wait_asynccnt(GATED ? 2 : 1);  // current panel done
#endif
        } else {
#if USE_ASYNC_LDS
            __builtin_amdgcn_s_wait_asynccnt(0);
#endif
        }
        __syncthreads();

        const u16* arow = A + (size_t)mRow * Kdim + it * 32;
        __builtin_prefetch(arow + 32, 0, 1);
        v16bf afrag = load16bf(arow + kHalf, arow + kHalf + 8);

        // Pre-load all B fragments (clause the ds_loads), then burst the WMMAs.
        v16bf bf1[4], bf2[4];
        #pragma unroll
        for (int nt = 0; nt < 4; ++nt) {
            const u16* bp = &lds[buf][0][(nt * 16 + (lane & 15)) * 32 + (lane >> 4) * 16];
            bf1[nt] = load16bf(bp, bp + 8);
            if constexpr (GATED) {
                const u16* bp2 = &lds[buf][1][(nt * 16 + (lane & 15)) * 32 + (lane >> 4) * 16];
                bf2[nt] = load16bf(bp2, bp2 + 8);
            }
        }
        #pragma unroll
        for (int nt = 0; nt < 4; ++nt) {
            acc1[nt] = __builtin_amdgcn_wmma_f32_16x16x32_bf16(
                false, afrag, false, bf1[nt], (short)0, acc1[nt], false, false);
            if constexpr (GATED) {
                acc2[nt] = __builtin_amdgcn_wmma_f32_16x16x32_bf16(
                    false, afrag, false, bf2[nt], (short)0, acc2[nt], false, false);
            }
        }
        __syncthreads();
    }

    // Epilogue. C layout: v8f element i -> row (lane/16)*8 + i, col lane%16.
    #pragma unroll
    for (int nt = 0; nt < 4; ++nt) {
        int col = nBase + nt * 16 + (lane & 15);
        float b1c = bias1 ? bias1[col] : 0.f;
        float b2c = 0.f;
        if constexpr (GATED) b2c = bias2 ? bias2[col] : 0.f;
        #pragma unroll
        for (int i = 0; i < 8; i++) {
            int row = mBase + wave * 16 + (lane >> 4) * 8 + i;
            size_t idx = (size_t)row * N + col;
            float a1 = acc1[nt][i] + b1c;
            if constexpr (GATED) {
                float a2 = acc2[nt][i] + b2c;
                float gv = (a1 / (1.f + __expf(-a1))) * a2;      // silu(a1)*a2
                if (epi == EPI_PHI)        outF[idx] = (gv > 0.f) ? (gv + 1.f) : __expf(gv); // elu+1
                else if (epi == EPI_GF32)  outF[idx] = gv;
                else                       outB[idx] = f2bf(gv);
            } else {
                if (epi == EPI_RESID)      outF[idx] = a1 + resid[idx];
                else if (epi == EPI_SACC)  outF[idx] += a1 * scale[(size_t)row * scaleStride];
                else                       outF[idx] = a1;
            }
        }
    }
}

// ---------------------------------------------------------------------------
// kv[b,h,d,e] = sum_s phi_k[b,s,h,d] * v[b,s,h,e] ;  ksum[b,h,d] = sum_s phi_k
// one block per (b,h); thread: d = t/4, e-range = (t%4)*16..+15
// ---------------------------------------------------------------------------
__global__ __launch_bounds__(256) void kv_ksum_kernel(const float* __restrict__ phik,
                                                      const float* __restrict__ vbuf,
                                                      float* __restrict__ kv,
                                                      float* __restrict__ ksum) {
    int bh = blockIdx.x; int b = bh >> 4; int h = bh & 15;
    int t = threadIdx.x; int d = t >> 2; int e0 = (t & 3) * 16;
    float acc[16]; float ks = 0.f;
    #pragma unroll
    for (int i = 0; i < 16; i++) acc[i] = 0.f;
    for (int s = 0; s < kS; ++s) {
        size_t base = ((size_t)(b * kS + s)) * kD + h * kDK;
        float pk = phik[base + d];
        ks += pk;
        const float* vp = vbuf + base + e0;
        #pragma unroll
        for (int i = 0; i < 16; i++) acc[i] += pk * vp[i];
    }
    size_t kb = ((size_t)bh * kDK + d) * kDK + e0;
    #pragma unroll
    for (int i = 0; i < 16; i++) kv[kb + i] = acc[i];
    if ((t & 3) == 0) ksum[(size_t)bh * kDK + d] = ks;
}

// ---------------------------------------------------------------------------
// attn[b,s,h*64+e] = (phi_q . kv) / (phi_q . ksum + eps), bf16 out
// one block per token; thread: h = t/16, e-range = (t%16)*4..+3
// ---------------------------------------------------------------------------
__global__ __launch_bounds__(256) void attn_kernel(const float* __restrict__ phiq,
                                                   const float* __restrict__ kv,
                                                   const float* __restrict__ ksum,
                                                   u16* __restrict__ attn) {
    int tok = blockIdx.x; int b = tok >> 11;           // tok / 2048
    int t = threadIdx.x; int h = t >> 4; int e0 = (t & 15) * 4;
    size_t base = (size_t)tok * kD + h * kDK;
    size_t kvb  = ((size_t)(b * kH + h)) * kDK * kDK;
    const float* ks = ksum + (size_t)(b * kH + h) * kDK;
    float acc[4] = {0.f, 0.f, 0.f, 0.f}; float qk = 0.f;
    for (int d = 0; d < kDK; ++d) {
        float pq = phiq[base + d];
        qk += pq * ks[d];
        const float* kp = kv + kvb + d * kDK + e0;
        #pragma unroll
        for (int i = 0; i < 4; i++) acc[i] += pq * kp[i];
    }
    float inv = 1.f / (qk + 1e-6f);
    #pragma unroll
    for (int i = 0; i < 4; i++) attn[base + e0 + i] = f2bf(acc[i] * inv);
}

// ---------------------------------------------------------------------------
// MoE gate: softmax(x2 @ gate_w + gate_b), top-2 renormalized -> comb[M][8]
// ---------------------------------------------------------------------------
__global__ __launch_bounds__(256) void gate_kernel(const u16* __restrict__ x2,
                                                   const float* __restrict__ gw,
                                                   const float* __restrict__ gb,
                                                   float* __restrict__ comb) {
    __shared__ float s[kE * 256];
    int row = blockIdx.x, t = threadIdx.x;
    float acc[kE];
    #pragma unroll
    for (int e = 0; e < kE; e++) acc[e] = 0.f;
    for (int d = t; d < kD; d += 256) {
        float xv = bf2f(x2[(size_t)row * kD + d]);
        const float* g = gw + (size_t)d * kE;
        #pragma unroll
        for (int e = 0; e < kE; e++) acc[e] += xv * g[e];
    }
    #pragma unroll
    for (int e = 0; e < kE; e++) s[e * 256 + t] = acc[e];
    __syncthreads();
    for (int st = 128; st > 0; st >>= 1) {
        if (t < st) {
            #pragma unroll
            for (int e = 0; e < kE; e++) s[e * 256 + t] += s[e * 256 + t + st];
        }
        __syncthreads();
    }
    if (t == 0) {
        float p[kE], mx = -1e30f;
        #pragma unroll
        for (int e = 0; e < kE; e++) { p[e] = s[e * 256] + gb[e]; mx = fmaxf(mx, p[e]); }
        float se = 0.f;
        #pragma unroll
        for (int e = 0; e < kE; e++) { p[e] = __expf(p[e] - mx); se += p[e]; }
        float invs = 1.f / se;
        #pragma unroll
        for (int e = 0; e < kE; e++) p[e] *= invs;
        int i0 = 0;
        for (int e = 1; e < kE; e++) if (p[e] > p[i0]) i0 = e;
        int i1 = -1;
        for (int e = 0; e < kE; e++) if (e != i0 && (i1 < 0 || p[e] > p[i1])) i1 = e;
        float inv2 = 1.f / (p[i0] + p[i1] + 1e-6f);
        #pragma unroll
        for (int e = 0; e < kE; e++)
            comb[(size_t)row * kE + e] = (e == i0) ? p[i0] * inv2 : (e == i1) ? p[i1] * inv2 : 0.f;
    }
}

// ---------------------------------------------------------------------------
// Host orchestration
// ---------------------------------------------------------------------------
extern "C" void kernel_launch(void* const* d_in, const int* in_sizes, int n_in,
                              void* d_out, int out_size, void* d_ws, size_t ws_size,
                              hipStream_t stream) {
    const float* x    = (const float*)d_in[0];
    const float* wq1  = (const float*)d_in[1];  const float* bq1 = (const float*)d_in[2];
    const float* wq2  = (const float*)d_in[3];  const float* bq2 = (const float*)d_in[4];
    const float* wk1  = (const float*)d_in[5];  const float* bk1 = (const float*)d_in[6];
    const float* wk2  = (const float*)d_in[7];  const float* bk2 = (const float*)d_in[8];
    const float* wv1  = (const float*)d_in[9];  const float* bv1 = (const float*)d_in[10];
    const float* wv2  = (const float*)d_in[11]; const float* bv2 = (const float*)d_in[12];
    const float* wo   = (const float*)d_in[13]; const float* bo  = (const float*)d_in[14];
    const float* ln1g = (const float*)d_in[15]; const float* ln1b = (const float*)d_in[16];
    const float* ln2g = (const float*)d_in[17]; const float* ln2b = (const float*)d_in[18];
    const float* gw   = (const float*)d_in[19]; const float* gb  = (const float*)d_in[20];
    const float* ew1  = (const float*)d_in[21]; const float* eb1 = (const float*)d_in[22];
    const float* ew2  = (const float*)d_in[23]; const float* eb2 = (const float*)d_in[24];
    const float* ew3  = (const float*)d_in[25]; const float* eb3 = (const float*)d_in[26];
    float* out = (float*)d_out;

    // workspace carve-out (256B aligned)
    char* ws = (char*)d_ws;
    size_t off = 0;
    auto take = [&](size_t bytes) -> char* {
        char* p = ws + off;
        off += (bytes + 255) & ~(size_t)255;
        return p;
    };
    u16* wq1T = (u16*)take((size_t)kD * kD * 2);
    u16* wq2T = (u16*)take((size_t)kD * kD * 2);
    u16* wk1T = (u16*)take((size_t)kD * kD * 2);
    u16* wk2T = (u16*)take((size_t)kD * kD * 2);
    u16* wv1T = (u16*)take((size_t)kD * kD * 2);
    u16* wv2T = (u16*)take((size_t)kD * kD * 2);
    u16* woT  = (u16*)take((size_t)kD * kD * 2);
    u16* e1T  = (u16*)take((size_t)kE * kD * kF * 2);  // per expert: [F][D]
    u16* e3T  = (u16*)take((size_t)kE * kD * kF * 2);  // per expert: [F][D]
    u16* e2T  = (u16*)take((size_t)kE * kF * kD * 2);  // per expert: [D][F]
    u16* x2   = (u16*)take((size_t)kM * kD * 2);
    float* phiq = (float*)take((size_t)kM * kD * 4);
    float* phik = (float*)take((size_t)kM * kD * 4);
    float* vbuf = (float*)take((size_t)kM * kD * 4);
    float* kv   = (float*)take((size_t)kB * kH * kDK * kDK * 4);
    float* ksum = (float*)take((size_t)kB * kH * kDK * 4);
    u16* attnb  = (u16*)take((size_t)kM * kD * 2);
    float* comb = (float*)take((size_t)kM * kE * 4);
    u16* hbuf   = (u16*)take((size_t)kM * kF * 2);

    // 1) pack all weights fp32 [K][N] -> bf16 [N][K]
    auto pack = [&](const float* in, u16* o, int K, int N) {
        packT<<<dim3(N / 32, K / 32), 256, 0, stream>>>(in, o, K, N);
    };
    pack(wq1, wq1T, kD, kD); pack(wq2, wq2T, kD, kD);
    pack(wk1, wk1T, kD, kD); pack(wk2, wk2T, kD, kD);
    pack(wv1, wv1T, kD, kD); pack(wv2, wv2T, kD, kD);
    pack(wo,  woT,  kD, kD);
    for (int e = 0; e < kE; e++) {
        pack(ew1 + (size_t)e * kD * kF, e1T + (size_t)e * kD * kF, kD, kF);
        pack(ew3 + (size_t)e * kD * kF, e3T + (size_t)e * kD * kF, kD, kF);
        pack(ew2 + (size_t)e * kF * kD, e2T + (size_t)e * kF * kD, kF, kD);
    }

    // 2) LN1 -> x2 (bf16)
    ln_kernel<<<kM, 256, 0, stream>>>(x, ln1g, ln1b, x2, 1e-5f);

    // 3) gated projections: phi_q, phi_k (elu+1 fused), v
    dim3 gD(kD / 64, kM / 128);
    gemm_wmma<true><<<gD, 256, 0, stream>>>(x2, kD, kD, wq1T, bq1, wq2T, bq2,
                                            phiq, nullptr, nullptr, nullptr, 0, EPI_PHI);
    gemm_wmma<true><<<gD, 256, 0, stream>>>(x2, kD, kD, wk1T, bk1, wk2T, bk2,
                                            phik, nullptr, nullptr, nullptr, 0, EPI_PHI);
    gemm_wmma<true><<<gD, 256, 0, stream>>>(x2, kD, kD, wv1T, bv1, wv2T, bv2,
                                            vbuf, nullptr, nullptr, nullptr, 0, EPI_GF32);

    // 4) linear attention reductions
    kv_ksum_kernel<<<kB * kH, 256, 0, stream>>>(phik, vbuf, kv, ksum);
    attn_kernel<<<kM, 256, 0, stream>>>(phiq, kv, ksum, attnb);

    // 5) output projection + residual -> d_out (fp32)
    gemm_wmma<false><<<gD, 256, 0, stream>>>(attnb, kD, kD, woT, bo, nullptr, nullptr,
                                             out, nullptr, x, nullptr, 0, EPI_RESID);

    // 6) LN2 -> x2 (bf16, reuse), gate -> comb
    ln_kernel<<<kM, 256, 0, stream>>>(out, ln2g, ln2b, x2, 1e-5f);
    gate_kernel<<<kM, 256, 0, stream>>>(x2, gw, gb, comb);

    // 7) dense experts (matches reference), scaled-accumulate into d_out
    dim3 gF(kF / 64, kM / 128);
    for (int e = 0; e < kE; e++) {
        gemm_wmma<true><<<gF, 256, 0, stream>>>(x2, kD, kF,
                                                e1T + (size_t)e * kD * kF, eb1 + (size_t)e * kF,
                                                e3T + (size_t)e * kD * kF, eb3 + (size_t)e * kF,
                                                nullptr, hbuf, nullptr, nullptr, 0, EPI_GBF16);
        gemm_wmma<false><<<gD, 256, 0, stream>>>(hbuf, kF, kD,
                                                 e2T + (size_t)e * kF * kD, eb2 + (size_t)e * kD,
                                                 nullptr, nullptr,
                                                 out, nullptr, nullptr,
                                                 comb + e, kE, EPI_SACC);
    }
    (void)in_sizes; (void)n_in; (void)out_size; (void)ws_size;
}
